// Brain_GNN_40853728920211
// MI455X (gfx1250) — matrix operand order repaired and checked by reading, not compiled, not run on hardware
//
#include <hip/hip_runtime.h>

typedef __attribute__((ext_vector_type(16))) _Float16 v16h;
typedef __attribute__((ext_vector_type(8)))  float    v8f;
typedef __attribute__((ext_vector_type(8)))  unsigned v8u;

// 16-bit A-matrix 16x32 layout: lane<16 -> K={0..7,16..23}, lane>=16 -> +8
__device__ __forceinline__ int a16_kmap(int i, int g) {
    const int p = i >> 1, q = i & 1;
    return (p < 4 ? (2 * p + q) : (16 + 2 * (p - 4) + q)) + 8 * g;
}

__device__ __forceinline__ float fast_tanh(float v) {
#if __has_builtin(__builtin_amdgcn_tanhf)
    return __builtin_amdgcn_tanhf(v);               // v_tanh_f32
#else
    return 1.0f - 2.0f * __builtin_amdgcn_rcpf(__expf(2.0f * v) + 1.0f);
#endif
}
// sigmoid(x) = 0.5*tanh(0.5x) + 0.5 : mul + v_tanh + fma, branch-free
__device__ __forceinline__ float fast_sigmoid(float v) {
    return fmaf(0.5f, fast_tanh(0.5f * v), 0.5f);
}
__device__ __forceinline__ unsigned pack_h2(float a, float b) {
    auto p = __builtin_amdgcn_cvt_pkrtz(a, b);       // v_cvt_pk_rtz_f16_f32
    return __builtin_bit_cast(unsigned, p);
}
// lane l -> lane l^16 exchange (SWAPX16): group-of-32 swizzle, xor_mask=0x10
__device__ __forceinline__ unsigned swap16(unsigned v) {
    return (unsigned)__builtin_amdgcn_ds_swizzle((int)v, 0x401F);
}

// ---------------------------------------------------------------------------
// GRU: each wave owns 16 nodes. Per step: hg^T = W_hh x h^T via 6 WMMAs
// (gate-dims 0..95 in six 16-row tiles), nodes along lanes.
// helem: he0[v] = h[node][v + 8g], he1[v] = h[node][16 + v + 8g].
// ---------------------------------------------------------------------------
__global__ __launch_bounds__(256) void gru_wmma_kernel(
    const float* __restrict__ x, const int* __restrict__ lengths,
    const int* __restrict__ batch,
    const float* __restrict__ W_ih, const float* __restrict__ W_hh,
    const float* __restrict__ b_ih, const float* __restrict__ b_hh,
    float* __restrict__ hout, int N, int T)
{
    const int lane = threadIdx.x & 31;
    const int gw   = blockIdx.x * 8 + (threadIdx.x >> 5);
    const int base = gw * 16;
    if (base >= N) return;                       // wave-uniform branch
    const int ln = lane & 15;
    const int g  = lane >> 4;
    const int node = base + ln;
    const int nlen = lengths[batch[node]];

    // wave max length -> loop bound (h frozen past every lane's length)
    int maxlen = nlen;
    #pragma unroll
    for (int s = 8; s >= 1; s >>= 1) {
        int o = __shfl_xor(maxlen, s, 32);
        maxlen = (o > maxlen) ? o : maxlen;
    }

    // A fragments: W_hh rows 16c+ln, K = i + 16g (contiguous)
    v16h a[6];
    #pragma unroll
    for (int c = 0; c < 6; ++c) {
        const int row = 16 * c + ln;
        #pragma unroll
        for (int i = 0; i < 16; ++i)
            a[c][i] = (_Float16)W_hh[row * 32 + i + 16 * g];
    }
    // C-init: r/z tiles get b_ih+b_hh, n tiles get b_hh only (r multiplies it)
    v8f cinit[6], wif[6], bihn[2];
    #pragma unroll
    for (int c = 0; c < 6; ++c) {
        #pragma unroll
        for (int v = 0; v < 8; ++v) {
            const int gd = 16 * c + v + 8 * g;
            cinit[c][v] = (c < 4) ? (b_ih[gd] + b_hh[gd]) : b_hh[gd];
            wif[c][v]   = W_ih[gd];
        }
    }
    #pragma unroll
    for (int tt = 0; tt < 2; ++tt)
        #pragma unroll
        for (int v = 0; v < 8; ++v)
            bihn[tt][v] = b_ih[64 + 16 * tt + v + 8 * g];

    v8f he0, he1;
    #pragma unroll
    for (int v = 0; v < 8; ++v) { he0[v] = 0.0f; he1[v] = 0.0f; }

    const float* xrow = x + (size_t)node * T;
    #pragma unroll
    for (int pb = 0; pb < 4; ++pb)                     // whole 1KB x row
        __builtin_prefetch(xrow + pb * 64, 0, 0);      // global_prefetch_b8

    float xt = xrow[0];
    for (int t = 0; t < maxlen; ++t) {
        // pipeline next x load past the WMMAs (clamped, branch-free)
        const int tn = (t + 1 < T) ? (t + 1) : (T - 1);
        const float xnext = xrow[tn];

        // Rebuild B fragment h^T. Static-index placement only:
        // g==0 keeps he0 words (elems 0..7), receives partner he0 (8..15);
        // g==1 keeps he1 words (elems 8..15), receives partner he1 (0..7).
        v8u hbw;
        #pragma unroll
        for (int v4 = 0; v4 < 4; ++v4) {
            const unsigned lo_w = pack_h2(he0[2 * v4], he0[2 * v4 + 1]);
            const unsigned hi_w = pack_h2(he1[2 * v4], he1[2 * v4 + 1]);
            const unsigned sw = g ? lo_w : hi_w;       // send own other-half
            const unsigned rw = swap16(sw);            // ds_swizzle SWAPX16
            hbw[v4]     = g ? rw : lo_w;
            hbw[4 + v4] = g ? hi_w : rw;
        }
        const v16h hb = __builtin_bit_cast(v16h, hbw);

        v8f d0 = __builtin_amdgcn_wmma_f32_16x16x32_f16(false, a[0], false, hb, (short)0, cinit[0], false, false);
        v8f d1 = __builtin_amdgcn_wmma_f32_16x16x32_f16(false, a[1], false, hb, (short)0, cinit[1], false, false);
        v8f d2 = __builtin_amdgcn_wmma_f32_16x16x32_f16(false, a[2], false, hb, (short)0, cinit[2], false, false);
        v8f d3 = __builtin_amdgcn_wmma_f32_16x16x32_f16(false, a[3], false, hb, (short)0, cinit[3], false, false);
        v8f d4 = __builtin_amdgcn_wmma_f32_16x16x32_f16(false, a[4], false, hb, (short)0, cinit[4], false, false);
        v8f d5 = __builtin_amdgcn_wmma_f32_16x16x32_f16(false, a[5], false, hb, (short)0, cinit[5], false, false);

        const bool upd = (t < nlen);
        #pragma unroll
        for (int v = 0; v < 8; ++v) {
            float r0 = fast_sigmoid(d0[v] + xt * wif[0][v]);
            float z0 = fast_sigmoid(d2[v] + xt * wif[2][v]);
            float n0 = fast_tanh(xt * wif[4][v] + bihn[0][v] + r0 * d4[v]);
            float h0 = n0 + z0 * (he0[v] - n0);
            he0[v] = upd ? h0 : he0[v];
            float r1 = fast_sigmoid(d1[v] + xt * wif[1][v]);
            float z1 = fast_sigmoid(d3[v] + xt * wif[3][v]);
            float n1 = fast_tanh(xt * wif[5][v] + bihn[1][v] + r1 * d5[v]);
            float h1 = n1 + z1 * (he1[v] - n1);
            he1[v] = upd ? h1 : he1[v];
        }
        xt = xnext;
    }
    #pragma unroll
    for (int v = 0; v < 8; ++v) {
        hout[(size_t)node * 32 + (v + 8 * g)]      = he0[v];
        hout[(size_t)node * 32 + (16 + v + 8 * g)] = he1[v];
    }
}

// ---------------------------------------------------------------------------
// GCN dense part: xw = act(hin) @ W^T (fused PReLU-on-load for layers >0),
// plus epilogue acc = xw * dis^2 + bias (the self term the scatter adds onto).
// ---------------------------------------------------------------------------
template <int DIN>
__global__ __launch_bounds__(256) void gcn_gemm_kernel(
    const float* __restrict__ hin, const float* __restrict__ alpha,
    const float* __restrict__ W, const float* __restrict__ bias,
    const float* __restrict__ dis, float* __restrict__ xw,
    float* __restrict__ acc, int N)
{
    const int lane = threadIdx.x & 31;
    const int gw   = blockIdx.x * 8 + (threadIdx.x >> 5);
    const int base = gw * 16;
    if (base >= N) return;
    const int ln = lane & 15;
    const int g  = lane >> 4;
    const float ap = alpha ? alpha[0] : 0.0f;
    constexpr int KB = DIN / 32;

    v16h afr[KB];
    const int arow = base + ln;
    #pragma unroll
    for (int kb = 0; kb < KB; ++kb) {
        #pragma unroll
        for (int i = 0; i < 16; ++i) {
            const int k = kb * 32 + i + 16 * g;
            float v = hin[(size_t)arow * DIN + k];
            if (alpha) v = (v >= 0.0f) ? v : ap * v;
            afr[kb][i] = (_Float16)v;
        }
    }
    v16h bfr[4][KB];
    #pragma unroll
    for (int c = 0; c < 4; ++c)
        #pragma unroll
        for (int kb = 0; kb < KB; ++kb)
            #pragma unroll
            for (int i = 0; i < 16; ++i) {
                const int k = kb * 32 + i + 16 * g;   // B: lane half splits K
                bfr[c][kb][i] = (_Float16)W[(size_t)(16 * c + ln) * DIN + k];
            }

    v8f d[4];
    #pragma unroll
    for (int c = 0; c < 4; ++c) {
        #pragma unroll
        for (int v = 0; v < 8; ++v) d[c][v] = 0.0f;
        #pragma unroll
        for (int kb = 0; kb < KB; ++kb)
            d[c] = __builtin_amdgcn_wmma_f32_16x16x32_f16(
                false, afr[kb], false, bfr[c][kb], (short)0, d[c], false, false);
    }

    float bcol[4];
    #pragma unroll
    for (int c = 0; c < 4; ++c) bcol[c] = bias[16 * c + ln];
    #pragma unroll
    for (int v = 0; v < 8; ++v) {
        const int node = base + v + 8 * g;
        const float ds = dis[node];
        const float d2 = ds * ds;
        #pragma unroll
        for (int c = 0; c < 4; ++c) {
            const size_t o = (size_t)node * 64 + 16 * c + ln;
            const float val = d[c][v];
            xw[o]  = val;
            acc[o] = val * d2 + bcol[c];
        }
    }
}

// ---------------------------------------------------------------------------
// Edge scatter: acc[dst] += xw[src] * coef. 4 feats/thread via b128 load;
// atomics land in L2 (acc is 1.9 MB, fully resident).
// ---------------------------------------------------------------------------
__global__ void gcn_scatter_kernel(const int* __restrict__ ei,
                                   const float* __restrict__ coef,
                                   const float* __restrict__ xw,
                                   float* __restrict__ acc, int E)
{
    const long long idx = (long long)blockIdx.x * blockDim.x + threadIdx.x;
    if (idx >= (long long)E * 16) return;
    const int f = (int)(idx & 15) * 4;
    const int e = (int)(idx >> 4);
    const int s  = ei[e];
    const int dd = ei[E + e];
    const float c = coef[e];
    const float4 v = *(const float4*)(xw + (size_t)s * 64 + f);
    float* ap = acc + (size_t)dd * 64 + f;
    atomicAdd(ap + 0, v.x * c);
    atomicAdd(ap + 1, v.y * c);
    atomicAdd(ap + 2, v.z * c);
    atomicAdd(ap + 3, v.w * c);
}

__global__ void deg_init_kernel(float* __restrict__ deg, int N) {
    const int i = blockIdx.x * blockDim.x + threadIdx.x;
    if (i < N) deg[i] = 1.0f;
}
__global__ void deg_accum_kernel(const int* __restrict__ ei,
                                 const float* __restrict__ ea,
                                 float* __restrict__ deg, int E) {
    const int e = blockIdx.x * blockDim.x + threadIdx.x;
    if (e < E) atomicAdd(&deg[ei[E + e]], ea[e]);
}
__global__ void deg_rsqrt_kernel(float* __restrict__ deg, int N) {
    const int i = blockIdx.x * blockDim.x + threadIdx.x;
    if (i < N) deg[i] = rsqrtf(deg[i]);
}
__global__ void coef_kernel(const int* __restrict__ ei,
                            const float* __restrict__ ea,
                            const float* __restrict__ dis,
                            float* __restrict__ coef, int E) {
    const int e = blockIdx.x * blockDim.x + threadIdx.x;
    if (e < E) coef[e] = dis[ei[e]] * ea[e] * dis[ei[E + e]];
}

// ---------------------------------------------------------------------------
// Finalize: PReLU(layer-3 accum), write node_emb, mean-pool per graph.
// ---------------------------------------------------------------------------
__global__ __launch_bounds__(256) void finalize_kernel(
    const float* __restrict__ acc, const float* __restrict__ prelu_a,
    float* __restrict__ node_emb, float* __restrict__ graph_emb,
    int nodes_per_graph)
{
    __shared__ float s[256];
    const int b = blockIdx.x;
    const int f = threadIdx.x & 63;
    const int r = threadIdx.x >> 6;
    const float a3 = prelu_a[3];
    float partial = 0.0f;
    for (int ni = r; ni < nodes_per_graph; ni += 4) {
        const size_t node = (size_t)b * nodes_per_graph + ni;
        float v = acc[node * 64 + f];
        v = (v >= 0.0f) ? v : a3 * v;
        node_emb[node * 64 + f] = v;
        partial += v;
    }
    s[threadIdx.x] = partial;
    __syncthreads();
    if (r == 0)
        graph_emb[(size_t)b * 64 + f] =
            (s[f] + s[64 + f] + s[128 + f] + s[192 + f]) / (float)nodes_per_graph;
}

extern "C" void kernel_launch(void* const* d_in, const int* in_sizes, int n_in,
                              void* d_out, int out_size, void* d_ws, size_t ws_size,
                              hipStream_t stream) {
    const float* x       = (const float*)d_in[0];
    const int*   ei      = (const int*)d_in[1];
    const float* ea      = (const float*)d_in[2];
    const int*   lengths = (const int*)d_in[3];
    const int*   batch   = (const int*)d_in[4];
    const float* W_ih    = (const float*)d_in[5];
    const float* W_hh    = (const float*)d_in[6];
    const float* b_ih    = (const float*)d_in[7];
    const float* b_hh    = (const float*)d_in[8];
    const float* g_w0    = (const float*)d_in[9];
    const float* g_b0    = (const float*)d_in[10];
    const float* g_ws    = (const float*)d_in[11];
    const float* g_bs    = (const float*)d_in[12];
    const float* prelu_a = (const float*)d_in[13];

    const int N  = in_sizes[4];          // batch array has N entries
    const int T  = in_sizes[0] / N;
    const int E  = in_sizes[2];
    const int B  = in_sizes[3];
    const int NPG = N / B;               // nodes per graph (116)

    size_t off = 0;
    auto wsalloc = [&](size_t bytes) {
        void* p = (char*)d_ws + off;
        off += (bytes + 255) & ~(size_t)255;
        return p;
    };
    float* dis  = (float*)wsalloc((size_t)N * 4);
    float* coef = (float*)wsalloc((size_t)E * 4);
    float* hgru = (float*)wsalloc((size_t)N * 32 * 4);
    float* xw   = (float*)wsalloc((size_t)N * 64 * 4);
    float* accA = (float*)wsalloc((size_t)N * 64 * 4);
    float* accB = (float*)wsalloc((size_t)N * 64 * 4);
    (void)ws_size; (void)n_in; (void)out_size;

    // symmetric normalization factors (deg -> rsqrt in place) + edge coef
    deg_init_kernel<<<(N + 255) / 256, 256, 0, stream>>>(dis, N);
    deg_accum_kernel<<<(E + 255) / 256, 256, 0, stream>>>(ei, ea, dis, E);
    deg_rsqrt_kernel<<<(N + 255) / 256, 256, 0, stream>>>(dis, N);
    coef_kernel<<<(E + 255) / 256, 256, 0, stream>>>(ei, ea, dis, coef, E);

    // GRU: 16 nodes / wave, 8 waves / block
    const int tileBlocks = (N / 16 + 7) / 8;
    gru_wmma_kernel<<<tileBlocks, 256, 0, stream>>>(x, lengths, batch, W_ih, W_hh,
                                                    b_ih, b_hh, hgru, N, T);

    const float* in = hgru;
    float* accs[2] = {accA, accB};
    for (int l = 0; l < 4; ++l) {
        const float* Wl = (l == 0) ? g_w0 : g_ws + (size_t)(l - 1) * 64 * 64;
        const float* bl = (l == 0) ? g_b0 : g_bs + (size_t)(l - 1) * 64;
        const float* al = (l == 0) ? nullptr : prelu_a + (l - 1);
        float* acc = accs[l & 1];
        if (l == 0)
            gcn_gemm_kernel<32><<<tileBlocks, 256, 0, stream>>>(in, al, Wl, bl, dis, xw, acc, N);
        else
            gcn_gemm_kernel<64><<<tileBlocks, 256, 0, stream>>>(in, al, Wl, bl, dis, xw, acc, N);
        const long long tot = (long long)E * 16;
        gcn_scatter_kernel<<<(unsigned)((tot + 255) / 256), 256, 0, stream>>>(ei, coef, xw, acc, E);
        in = acc;
    }

    float* node_emb  = (float*)d_out;
    float* graph_emb = node_emb + (size_t)N * 64;
    finalize_kernel<<<B, 256, 0, stream>>>(in, prelu_a, node_emb, graph_emb, NPG);
}
